// SwinTransformerBlock3D_17884243821113
// MI455X (gfx1250) — matrix operand order repaired and checked by reading, not compiled, not run on hardware
//
#include <hip/hip_runtime.h>
#include <math.h>

typedef _Float16 half_t;
typedef __attribute__((ext_vector_type(16))) _Float16 v16h;
typedef __attribute__((ext_vector_type(8)))  _Float16 v8h;
typedef __attribute__((ext_vector_type(8)))  float    v8f;

union AB16 { v16h v; v8h h[2]; };

#define DIMC    384
#define NHEADS  12
#define HD      32
#define NTOK    98
#define NWIN    1024
#define TOKENS  100352   // 4*8*56*56 == 1024*98 == 784*128 (multiple of 128)

// lane-XOR swizzle within wave32 halves (group-of-32 mode, and_mask=0x1f)
#define SWZ_XOR(x, m) \
    __int_as_float(__builtin_amdgcn_ds_swizzle(__float_as_int(x), ((m) << 10) | 0x1f))

// ---------------------------------------------------------------------------
// f32 -> f16 weight conversion
// ---------------------------------------------------------------------------
__global__ void cvt_f32_f16(const float* __restrict__ src, half_t* __restrict__ dst, int n) {
    int i = blockIdx.x * 256 + threadIdx.x;
    if (i < n) dst[i] = (half_t)src[i];
}

// ---------------------------------------------------------------------------
// relative position bias -> padded [12][112][112] f32 table (pad = 0)
// ---------------------------------------------------------------------------
__global__ void build_bias112(const float* __restrict__ table, const int* __restrict__ ridx,
                              float* __restrict__ bias112) {
    int i = blockIdx.x * 256 + threadIdx.x;
    if (i >= NHEADS * 112 * 112) return;
    int h  = i / (112 * 112);
    int rc = i % (112 * 112);
    int r = rc / 112, c = rc % 112;
    float b = 0.f;
    if (r < NTOK && c < NTOK) b = table[ridx[r * NTOK + c] * NHEADS + h];
    bias112[i] = b;
}

// ---------------------------------------------------------------------------
// attention mask -> padded [256][112][112] f32 table (pad = -1e30 so padded
// key columns vanish in softmax; keeps the attention epilogue branch-free)
// ---------------------------------------------------------------------------
__global__ void build_mask112(const float* __restrict__ mask, float* __restrict__ mask112) {
    long i = (long)blockIdx.x * 256 + threadIdx.x;
    if (i >= (long)256 * 112 * 112) return;
    int wb = (int)(i / (112 * 112));
    int rc = (int)(i % (112 * 112));
    int r = rc / 112, c = rc % 112;
    float m = -1.0e30f;
    if (r < NTOK && c < NTOK) m = mask[(long)wb * NTOK * NTOK + r * NTOK + c];
    mask112[i] = m;
}

// ---------------------------------------------------------------------------
// LayerNorm (+ optional shifted-window partition) -> f16 rows
// ---------------------------------------------------------------------------
__global__ void __launch_bounds__(128)
ln_kernel(const float* __restrict__ xin, const float* __restrict__ g,
          const float* __restrict__ b, half_t* __restrict__ out, int partition) {
    __shared__ float red[128];
    int tkn = blockIdx.x;
    long srcRow;
    if (partition) {
        int w = tkn / NTOK, n = tkn % NTOK;
        int bb = w >> 8, rem = w & 255;
        int wd = rem >> 6, rem2 = rem & 63;
        int wh = rem2 >> 3, ww2 = rem2 & 7;
        int td = n / 49, r3 = n % 49, th = r3 / 7, tw = r3 % 7;
        int dd = (wd * 2 + td + 1) & 7;                    // roll -1 over D=8
        int hh = wh * 7 + th + 3; if (hh >= 56) hh -= 56;  // roll -3 over H=56
        int wc = ww2 * 7 + tw + 3; if (wc >= 56) wc -= 56; // roll -3 over W=56
        srcRow = (long)(((bb * 8 + dd) * 56 + hh) * 56 + wc) * DIMC;
    } else {
        srcRow = (long)tkn * DIMC;
    }
    int t = threadIdx.x;
    float v0 = xin[srcRow + t];
    float v1 = xin[srcRow + t + 128];
    float v2 = xin[srcRow + t + 256];
    red[t] = v0 + v1 + v2;
    __syncthreads();
    for (int off = 64; off > 0; off >>= 1) { if (t < off) red[t] += red[t + off]; __syncthreads(); }
    float mu = red[0] * (1.f / DIMC);
    __syncthreads();
    float d0 = v0 - mu, d1 = v1 - mu, d2 = v2 - mu;
    red[t] = d0 * d0 + d1 * d1 + d2 * d2;
    __syncthreads();
    for (int off = 64; off > 0; off >>= 1) { if (t < off) red[t] += red[t + off]; __syncthreads(); }
    float rstd = rsqrtf(red[0] * (1.f / DIMC) + 1e-5f);
    long dstRow = (long)tkn * DIMC;
    out[dstRow + t]       = (half_t)(d0 * rstd * g[t]       + b[t]);
    out[dstRow + t + 128] = (half_t)(d1 * rstd * g[t + 128] + b[t + 128]);
    out[dstRow + t + 256] = (half_t)(d2 * rstd * g[t + 256] + b[t + 256]);
}

// ---------------------------------------------------------------------------
// Generic tiled WMMA GEMM: C[M,N] = A[M,K] * W[N,K]^T + bias, with epilogues.
// Block 256 threads (8 waves), tile 128x128, BK=32, double-buffered LDS.
// M, N, K must all be multiples of 128/128/32 (true for every call here).
// ---------------------------------------------------------------------------
#define EPI_F16      0
#define EPI_GELU_F16 1
#define EPI_PROJ_RES 2
#define EPI_FC2_RES  3

template<int EPI>
__global__ void __launch_bounds__(256)
gemm_wmma(const half_t* __restrict__ A, const half_t* __restrict__ W,
          const float* __restrict__ bias, int M, int N, int K,
          half_t* __restrict__ outH, float* __restrict__ outF,
          const float* __restrict__ res) {
    __shared__ __attribute__((aligned(32))) half_t lA[2][128 * 40];
    __shared__ __attribute__((aligned(32))) half_t lB[2][128 * 40];

    const int tid  = threadIdx.x;
    const int wave = tid >> 5, lane = tid & 31;
    const int lo = lane & 15, hi = lane >> 4;
    const int wm = wave & 3, wn = wave >> 2;
    const int mo = wm * 32, no = wn * 64;
    const long mBase = (long)blockIdx.x * 128;
    const int  nBase = blockIdx.y * 128;

    v8f acc[2][4];
#pragma unroll
    for (int i = 0; i < 2; i++)
#pragma unroll
        for (int j = 0; j < 4; j++) acc[i][j] = (v8f){};

    const int rr = tid >> 1, seg = tid & 1;  // staging: row 0..127, 32B segment

    const half_t* gA = A + (mBase + rr) * (long)K + seg * 16;
    const half_t* gW = W + ((long)nBase + rr) * (long)K + seg * 16;
    half_t* const sA = (half_t*)lA[0] + rr * 40 + seg * 16;
    half_t* const sB = (half_t*)lB[0] + rr * 40 + seg * 16;
    const int bufStride = 128 * 40;

    // prologue: stage k-step 0 into buffer 0
    {
        const uint4* pa = (const uint4*)gA;
        const uint4* pb = (const uint4*)gW;
        uint4* qa = (uint4*)sA;
        uint4* qb = (uint4*)sB;
        qa[0] = pa[0]; qa[1] = pa[1];
        qb[0] = pb[0]; qb[1] = pb[1];
    }
    __syncthreads();

    const int nk = K >> 5;
    for (int s = 0; s < nk; s++) {
        const int cur = s & 1;
        // stage next k-step into the other buffer (overlaps with WMMA below)
        if (s + 1 < nk) {
            const uint4* pa = (const uint4*)(gA + (s + 1) * 32);
            const uint4* pb = (const uint4*)(gW + (s + 1) * 32);
            uint4* qa = (uint4*)(sA + (cur ^ 1) * bufStride);
            uint4* qb = (uint4*)(sB + (cur ^ 1) * bufStride);
            qa[0] = pa[0]; qa[1] = pa[1];
            qb[0] = pb[0]; qb[1] = pb[1];
        }
        const half_t* cA = (const half_t*)lA[cur];
        const half_t* cB = (const half_t*)lB[cur];
        AB16 af[2], bf[4];
#pragma unroll
        for (int i = 0; i < 2; i++) {
            int row = mo + i * 16 + lo;
            af[i].h[0] = *(const v8h*)(cA + row * 40 + 16 * hi);
            af[i].h[1] = *(const v8h*)(cA + row * 40 + 16 * hi + 8);
        }
#pragma unroll
        for (int j = 0; j < 4; j++) {
            int col = no + j * 16 + lo;
            bf[j].h[0] = *(const v8h*)(cB + col * 40 + 16 * hi);
            bf[j].h[1] = *(const v8h*)(cB + col * 40 + 16 * hi + 8);
        }
#pragma unroll
        for (int i = 0; i < 2; i++)
#pragma unroll
            for (int j = 0; j < 4; j++)
                acc[i][j] = __builtin_amdgcn_wmma_f32_16x16x32_f16(
                    false, af[i].v, false, bf[j].v, (short)0, acc[i][j], false, false);
        __syncthreads();
    }

    // epilogue: lane holds (M = base + r + 8*hi, N = base + lo) per ISA C layout
#pragma unroll
    for (int i = 0; i < 2; i++) {
#pragma unroll
        for (int j = 0; j < 4; j++) {
#pragma unroll
            for (int r = 0; r < 8; r++) {
                long row = mBase + mo + i * 16 + r + 8 * hi;
                int  col = nBase + no + j * 16 + lo;
                float v = acc[i][j][r] + bias[col];
                if (EPI == EPI_F16) {
                    outH[row * (long)N + col] = (half_t)v;
                } else if (EPI == EPI_GELU_F16) {
                    float gl = 0.5f * v * (1.f + erff(v * 0.70710678118654752f));
                    outH[row * (long)N + col] = (half_t)gl;
                } else if (EPI == EPI_FC2_RES) {
                    outF[row * (long)N + col] = res[row * (long)N + col] + v;
                } else { // EPI_PROJ_RES: window-reverse + un-shift + residual
                    int tkn = (int)row;
                    int w = tkn / NTOK, n = tkn % NTOK;
                    int bb = w >> 8, rem = w & 255;
                    int wd = rem >> 6, rem2 = rem & 63;
                    int wh = rem2 >> 3, ww2 = rem2 & 7;
                    int td = n / 49, r3 = n % 49, th = r3 / 7, tw = r3 % 7;
                    int dd = (wd * 2 + td + 1) & 7;
                    int hh = wh * 7 + th + 3; if (hh >= 56) hh -= 56;
                    int wc = ww2 * 7 + tw + 3; if (wc >= 56) wc -= 56;
                    long dst = (long)(((bb * 8 + dd) * 56 + hh) * 56 + wc) * DIMC + col;
                    outF[dst] = res[dst] + v;
                }
            }
        }
    }
}

// ---------------------------------------------------------------------------
// Attention: one block per (window, head). Waves 0..6 own one 16-row M-tile of
// the 98(->112) tokens. QK^T (one 16x16x32 wmma per tile), branch-free fused
// scale + rel-bias + padded-mask + softmax (ds_swizzle XOR reductions, fully
// unrolled so accumulator access is constant-indexed), then P @ V.
// ---------------------------------------------------------------------------
__global__ void __launch_bounds__(256)
attn_wmma(const half_t* __restrict__ qkv, const float* __restrict__ mask112,
          const float* __restrict__ bias112, half_t* __restrict__ outA) {
    __shared__ __attribute__((aligned(32))) half_t lQ[112 * 32];
    __shared__ __attribute__((aligned(32))) half_t lK[112 * 32];
    __shared__ __attribute__((aligned(32))) half_t lVt[32 * 128];
    __shared__ __attribute__((aligned(32))) half_t lP[112 * 128];

    const int w = blockIdx.x / NHEADS;
    const int h = blockIdx.x % NHEADS;
    const int tid = threadIdx.x;
    const long qbase = (long)w * NTOK * 1152 + h * HD;  // halves; q +0, k +384, v +768

    // stage Q, K (zero-padded to 112 rows), V transposed (zero-padded to 128)
    {
        const unsigned* src = (const unsigned*)qkv;
        unsigned* dq = (unsigned*)lQ;
        unsigned* dk = (unsigned*)lK;
#pragma unroll
        for (int it = 0; it < 7; it++) {
            int i = it * 256 + tid;
            int row = i >> 4, dc = i & 15;
            unsigned vq = 0, vk = 0;
            if (row < NTOK) {
                long dw = (qbase + (long)row * 1152) >> 1;
                vq = src[dw + dc];
                vk = src[dw + 192 + dc];     // +384 halves
            }
            dq[i] = vq; dk[i] = vk;
        }
#pragma unroll
        for (int it = 0; it < 16; it++) {
            int i = it * 256 + tid;
            int d = i >> 7, tok = i & 127;
            half_t v = (half_t)0.f;
            if (tok < NTOK) v = qkv[qbase + (long)tok * 1152 + 768 + d];
            lVt[d * 128 + tok] = v;
        }
    }
    __syncthreads();

    const int wave = tid >> 5, lane = tid & 31;
    const int lo = lane & 15, hi = lane >> 4;
    if (wave < 7) {
        const int t = wave;  // M-tile: token rows 16t .. 16t+15
        AB16 aq;
        {
            int row = t * 16 + lo;
            aq.h[0] = *(const v8h*)(lQ + row * 32 + 16 * hi);
            aq.h[1] = *(const v8h*)(lQ + row * 32 + 16 * hi + 8);
        }
        v8f zero = {};
        v8f s[7];
#pragma unroll
        for (int nt = 0; nt < 7; nt++) {
            AB16 bk;
            int tok = nt * 16 + lo;
            bk.h[0] = *(const v8h*)(lK + tok * 32 + 16 * hi);
            bk.h[1] = *(const v8h*)(lK + tok * 32 + 16 * hi + 8);
            s[nt] = __builtin_amdgcn_wmma_f32_16x16x32_f16(
                false, aq.v, false, bk.v, (short)0, zero, false, false);
        }

        const float scale = 0.17677669529663687f;  // 32^-0.5
        const float* mrow = mask112 + (long)(w & 255) * 112 * 112;
        const float* brow = bias112 + h * 112 * 112;

        // branch-free softmax per row; fully unrolled -> constant vector idx
#pragma unroll
        for (int r = 0; r < 8; r++) {
            int row = t * 16 + r + 8 * hi;
            float tv[7];
            float mx = -3.0e38f;
#pragma unroll
            for (int nt = 0; nt < 7; nt++) {
                int col = nt * 16 + lo;
                float v = s[nt][r] * scale + brow[row * 112 + col] + mrow[row * 112 + col];
                tv[nt] = v;
                mx = fmaxf(mx, v);
            }
            mx = fmaxf(mx, SWZ_XOR(mx, 1));
            mx = fmaxf(mx, SWZ_XOR(mx, 2));
            mx = fmaxf(mx, SWZ_XOR(mx, 4));
            mx = fmaxf(mx, SWZ_XOR(mx, 8));
            float sum = 0.f;
            float ev[7];
#pragma unroll
            for (int nt = 0; nt < 7; nt++) { ev[nt] = __expf(tv[nt] - mx); sum += ev[nt]; }
            sum += SWZ_XOR(sum, 1);
            sum += SWZ_XOR(sum, 2);
            sum += SWZ_XOR(sum, 4);
            sum += SWZ_XOR(sum, 8);
            float rs = 1.f / sum;
#pragma unroll
            for (int nt = 0; nt < 7; nt++)
                lP[row * 128 + nt * 16 + lo] = (half_t)(ev[nt] * rs);
            lP[row * 128 + 112 + lo] = (half_t)0.f;   // zero K-pad cols 112..127
        }

        // O = P @ V   (K padded to 128, N = 32 head dims = 2 n-tiles)
        v8f o[2]; o[0] = (v8f){}; o[1] = (v8f){};
#pragma unroll
        for (int kk = 0; kk < 4; kk++) {
            AB16 ap;
            int row = t * 16 + lo;
            ap.h[0] = *(const v8h*)(lP + row * 128 + kk * 32 + 16 * hi);
            ap.h[1] = *(const v8h*)(lP + row * 128 + kk * 32 + 16 * hi + 8);
#pragma unroll
            for (int j = 0; j < 2; j++) {
                AB16 bv;
                int d = j * 16 + lo;
                bv.h[0] = *(const v8h*)(lVt + d * 128 + kk * 32 + 16 * hi);
                bv.h[1] = *(const v8h*)(lVt + d * 128 + kk * 32 + 16 * hi + 8);
                o[j] = __builtin_amdgcn_wmma_f32_16x16x32_f16(
                    false, ap.v, false, bv.v, (short)0, o[j], false, false);
            }
        }
#pragma unroll
        for (int j = 0; j < 2; j++)
#pragma unroll
            for (int r = 0; r < 8; r++) {
                int row = t * 16 + r + 8 * hi;
                if (row < NTOK) {
                    int d = j * 16 + lo;
                    outA[((long)w * NTOK + row) * DIMC + h * HD + d] = (half_t)o[j][r];
                }
            }
    }
}

// ---------------------------------------------------------------------------
// launcher
// ---------------------------------------------------------------------------
extern "C" void kernel_launch(void* const* d_in, const int* in_sizes, int n_in,
                              void* d_out, int out_size, void* d_ws, size_t ws_size,
                              hipStream_t stream) {
    const float* x      = (const float*)d_in[0];
    const float* maskp  = (const float*)d_in[1];
    const float* rtab   = (const float*)d_in[2];
    const int*   ridx   = (const int*)  d_in[3];
    const float* qkv_w  = (const float*)d_in[4];
    const float* qkv_b  = (const float*)d_in[5];
    const float* proj_w = (const float*)d_in[6];
    const float* proj_b = (const float*)d_in[7];
    const float* g1     = (const float*)d_in[8];
    const float* b1     = (const float*)d_in[9];
    const float* g2     = (const float*)d_in[10];
    const float* b2     = (const float*)d_in[11];
    const float* fc1_w  = (const float*)d_in[12];
    const float* fc1_b  = (const float*)d_in[13];
    const float* fc2_w  = (const float*)d_in[14];
    const float* fc2_b  = (const float*)d_in[15];

    char* ws = (char*)d_ws;
    size_t off = 0;
    auto take = [&](size_t bytes) {
        off = (off + 255) & ~(size_t)255;
        void* p = ws + off;
        off += bytes;
        return p;
    };
    half_t* wqkv16  = (half_t*)take((size_t)1152 * 384 * 2);
    half_t* wproj16 = (half_t*)take((size_t)384 * 384 * 2);
    half_t* wfc116  = (half_t*)take((size_t)384 * 384 * 2);
    half_t* wfc216  = (half_t*)take((size_t)384 * 384 * 2);
    float*  bias112 = (float*) take((size_t)NHEADS * 112 * 112 * 4);
    float*  mask112 = (float*) take((size_t)256 * 112 * 112 * 4);
    half_t* xw16    = (half_t*)take((size_t)TOKENS * 384 * 2);   // reused as h2
    half_t* qkv16   = (half_t*)take((size_t)TOKENS * 1152 * 2);  // reused as mlp1
    half_t* attn16  = (half_t*)take((size_t)TOKENS * 384 * 2);
    float*  x1      = (float*) take((size_t)TOKENS * 384 * 4);
    (void)ws_size; (void)in_sizes; (void)n_in; (void)out_size;

    // 1) weights -> f16
    cvt_f32_f16<<<(1152 * 384 + 255) / 256, 256, 0, stream>>>(qkv_w, wqkv16, 1152 * 384);
    cvt_f32_f16<<<(384 * 384 + 255) / 256, 256, 0, stream>>>(proj_w, wproj16, 384 * 384);
    cvt_f32_f16<<<(384 * 384 + 255) / 256, 256, 0, stream>>>(fc1_w, wfc116, 384 * 384);
    cvt_f32_f16<<<(384 * 384 + 255) / 256, 256, 0, stream>>>(fc2_w, wfc216, 384 * 384);

    // 2) padded bias + mask tables
    build_bias112<<<(NHEADS * 112 * 112 + 255) / 256, 256, 0, stream>>>(rtab, ridx, bias112);
    build_mask112<<<(int)(((long)256 * 112 * 112 + 255) / 256), 256, 0, stream>>>(maskp, mask112);

    // 3) LN1 + shifted-window partition -> f16 tokens
    ln_kernel<<<TOKENS, 128, 0, stream>>>(x, g1, b1, xw16, 1);

    // 4) QKV GEMM (100352 x 1152 x 384)
    gemm_wmma<EPI_F16><<<dim3(TOKENS / 128, 1152 / 128), 256, 0, stream>>>(
        xw16, wqkv16, qkv_b, TOKENS, 1152, 384, qkv16, nullptr, nullptr);

    // 5) windowed attention
    attn_wmma<<<NWIN * NHEADS, 256, 0, stream>>>(qkv16, mask112, bias112, attn16);

    // 6) proj GEMM + window-reverse + un-shift + residual -> x1 (f32)
    gemm_wmma<EPI_PROJ_RES><<<dim3(TOKENS / 128, 384 / 128), 256, 0, stream>>>(
        attn16, wproj16, proj_b, TOKENS, 384, 384, nullptr, x1, x);

    // 7) LN2 -> f16 (reuse xw16)
    ln_kernel<<<TOKENS, 128, 0, stream>>>(x1, g2, b2, xw16, 0);

    // 8) FC1 GEMM + exact GELU -> f16 (reuse qkv16)
    gemm_wmma<EPI_GELU_F16><<<dim3(TOKENS / 128, 384 / 128), 256, 0, stream>>>(
        xw16, wfc116, fc1_b, TOKENS, 384, 384, qkv16, nullptr, nullptr);

    // 9) FC2 GEMM + residual with x1 -> d_out (f32)
    gemm_wmma<EPI_FC2_RES><<<dim3(TOKENS / 128, 384 / 128), 256, 0, stream>>>(
        qkv16, wfc216, fc2_b, TOKENS, 384, 384, nullptr, (float*)d_out, x1);
}
